// GroupedQueryAttention_86517821216200
// MI455X (gfx1250) — matrix operand order repaired and checked by reading, not compiled, not run on hardware
//
#include <hip/hip_runtime.h>
#include <hip/hip_bf16.h>

// ---------------------------------------------------------------------------
// GQA attention layer for MI455X (gfx1250, wave32).
// All four GEMM stages + both attention matmuls use v_wmma_f32_16x16x32_bf16
// (bf16 inputs, fp32 accumulation). Compute-bound problem (~310 GFLOP vs
// ~9us of HBM traffic at 23.3 TB/s), so everything routes through WMMA.
// Attention Q/K tiles are staged into LDS via the Tensor Data Mover
// (tensor_load_to_lds + s_wait_tensorcnt) when the builtin is available.
// ---------------------------------------------------------------------------

typedef __attribute__((ext_vector_type(16))) __bf16 v16bf;
typedef __attribute__((ext_vector_type(8)))  float  v8f;
typedef __attribute__((ext_vector_type(4)))  unsigned u32x4;
typedef __attribute__((ext_vector_type(8)))  int      i32x8;
typedef __attribute__((ext_vector_type(4)))  int      i32x4;

#define NUM_HEADS   32
#define NUM_KV      8
#define HEAD_DIM    128
#define HSZ         4096
#define KVSZ        1024

#ifndef __has_builtin
#define __has_builtin(x) 0
#endif
#if __has_builtin(__builtin_amdgcn_tensor_load_to_lds)
#define HAVE_TDM 1
#endif

union Frag { v16bf v; unsigned u[8]; };

__device__ __forceinline__ unsigned short f2bf(float f) {
    unsigned u = __float_as_uint(f);
    u += 0x7FFFu + ((u >> 16) & 1u);   // round-to-nearest-even
    return (unsigned short)(u >> 16);
}
__device__ __forceinline__ float bf2f(unsigned short h) {
    return __uint_as_float(((unsigned)h) << 16);
}

// A-operand (16x32, 16-bit) K index for VGPR j, lane-half hi  (ISA 7.12.2)
__device__ __forceinline__ int kkA(int j, int hi) {
    return ((j & 3) << 1) + ((j >> 2) << 4) + (hi << 3);
}
// B-operand (32x16, 16-bit) K index for VGPR j, lane-half hi
__device__ __forceinline__ int kkB(int j, int hi) {
    return ((j & 3) << 1) + ((j >> 2) << 4) + (hi << 3);
}

__device__ __forceinline__ void wait_tensorcnt0() {
#if __has_builtin(__builtin_amdgcn_s_wait_tensorcnt)
    __builtin_amdgcn_s_wait_tensorcnt(0);
#else
    asm volatile("s_wait_tensorcnt 0x0" ::: "memory");
#endif
}

#ifdef HAVE_TDM
// ---------------------------------------------------------------------------
// TDM 2D tile load: tile_cols x tile_rows elements (2-byte), row stride
// stride_elems, global -> LDS (contiguous). D# per CDNA5 ISA 8.3/8.4.
// Issue from one wave with uniform operands; EXEC is ignored by TDM.
// ---------------------------------------------------------------------------
__device__ __forceinline__ void tdm_load_2d_bf16(
    unsigned lds_byte_off, const void* gptr,
    unsigned tile_cols, unsigned tile_rows, unsigned stride_elems)
{
    unsigned long long ga = (unsigned long long)(uintptr_t)gptr;
    const unsigned td0 = 0x40000000u, td1 = 0x40000000u;  // huge: no OOB clamp
    const unsigned long long st0 = (unsigned long long)stride_elems;

    u32x4 g0;
    g0.x = 1u;                                     // count=1 (valid user D#)
    g0.y = lds_byte_off;                           // lds_addr
    g0.z = (unsigned)(ga & 0xFFFFFFFFu);           // global_addr[31:0]
    g0.w = (unsigned)((ga >> 32) & 0x01FFFFFFu)    // global_addr[56:32]
         | (2u << 30);                             // type = 2 ("image")

    i32x8 g1;
    g1[0] = (int)(1u << 16);                                       // data_size=1 (2B)
    g1[1] = (int)((td0 & 0xFFFFu) << 16);                          // tensor_dim0[15:0]
    g1[2] = (int)(((td0 >> 16) & 0xFFFFu) | ((td1 & 0xFFFFu) << 16));
    g1[3] = (int)(((td1 >> 16) & 0xFFFFu) | (tile_cols << 16));    // tile_dim0
    g1[4] = (int)(tile_rows & 0xFFFFu);                            // tile_dim1, tile_dim2=0
    g1[5] = (int)(unsigned)(st0 & 0xFFFFFFFFu);                    // dim0_stride[31:0]
    g1[6] = (int)((st0 >> 32) & 0xFFFFu);                          // dim0_stride[47:32]
    g1[7] = 0;

    i32x4 gz = {0, 0, 0, 0};                                       // 2D: groups 2/3 unused
#if __clang_major__ >= 23
    i32x8 gz8 = {0, 0, 0, 0, 0, 0, 0, 0};
    __builtin_amdgcn_tensor_load_to_lds(g0, g1, gz, gz, gz8, 0);
#else
    __builtin_amdgcn_tensor_load_to_lds(g0, g1, gz, gz, 0);
#endif
}
#endif // HAVE_TDM

// ---------------------------------------------------------------------------
// fp32 -> bf16 elementwise convert
// ---------------------------------------------------------------------------
__global__ void cvt_f32_bf16(const float* __restrict__ src,
                             unsigned short* __restrict__ dst, int n) {
    int i = blockIdx.x * blockDim.x + threadIdx.x;
    if (i < n) dst[i] = f2bf(src[i]);
}

// ---------------------------------------------------------------------------
// GEMM: C[M,N] = A[M,K](bf16,row-major) x B[K,N](bf16,row-major)
// Block 256 thr = 8 waves; block tile 64(M) x 128(N); wave tile 16x64.
// ---------------------------------------------------------------------------
template <bool BF16OUT>
__global__ __launch_bounds__(256) void gemm_bf16(
    const unsigned short* __restrict__ A,
    const unsigned short* __restrict__ B,
    void* __restrict__ Cout, int M, int N, int K)
{
    __shared__ __align__(16) unsigned short As[64 * 32];    // [m][k]
    __shared__ __align__(16) unsigned short Bts[128 * 32];  // [n][k] (transposed)

    const int tid  = threadIdx.x;
    const int wave = tid >> 5;
    const int lane = tid & 31;
    const int hi   = lane >> 4;
    const int l16  = lane & 15;
    const int wm   = wave >> 1;          // 0..3
    const int wn   = wave & 1;           // 0..1
    const int m0   = blockIdx.y * 64;
    const int n0   = blockIdx.x * 128;

    v8f acc[4] = {};

    for (int k0 = 0; k0 < K; k0 += 32) {
        // --- stage A tile 64x32 (8 elems / thread, dword copies) ---
        {
            int e = tid * 8;
            int r = e >> 5, c = e & 31;
            const unsigned* ap = (const unsigned*)(A + (size_t)(m0 + r) * K + k0 + c);
            unsigned* dp = (unsigned*)(As + r * 32 + c);
            dp[0] = ap[0]; dp[1] = ap[1]; dp[2] = ap[2]; dp[3] = ap[3];
            if (k0 + 32 < K)
                __builtin_prefetch(A + (size_t)(m0 + r) * K + k0 + 32 + c, 0, 1);
        }
        // --- stage B tile 32x128 transposed into Bts[n][k] ---
        {
            int e = tid * 16;
            int r = e >> 7, c = e & 127;     // r = k row, c = n col base
            const unsigned* bp = (const unsigned*)(B + (size_t)(k0 + r) * N + n0 + c);
#pragma unroll
            for (int j = 0; j < 8; ++j) {
                unsigned d = bp[j];
                int cc = c + 2 * j;
                Bts[cc * 32 + r]       = (unsigned short)(d & 0xFFFFu);
                Bts[(cc + 1) * 32 + r] = (unsigned short)(d >> 16);
            }
            if (k0 + 32 < K)
                __builtin_prefetch(B + (size_t)(k0 + 32 + r) * N + n0 + c, 0, 1);
        }
        __syncthreads();

        Frag af;
#pragma unroll
        for (int j = 0; j < 8; ++j)
            af.u[j] = *(const unsigned*)(As + (wm * 16 + l16) * 32 + kkA(j, hi));

#pragma unroll
        for (int t = 0; t < 4; ++t) {
            Frag bf_;
#pragma unroll
            for (int j = 0; j < 8; ++j)
                bf_.u[j] = *(const unsigned*)(Bts + (wn * 64 + t * 16 + l16) * 32 + kkB(j, hi));
            acc[t] = __builtin_amdgcn_wmma_f32_16x16x32_bf16(
                false, af.v, false, bf_.v, (short)0, acc[t], false, false);
        }
        __syncthreads();
    }

    // --- epilogue: C layout = VGPR i -> row i + hi*8, lane&15 -> col ---
#pragma unroll
    for (int t = 0; t < 4; ++t) {
#pragma unroll
        for (int i = 0; i < 8; ++i) {
            int row = m0 + wm * 16 + i + hi * 8;
            int col = n0 + wn * 64 + t * 16 + l16;
            if (BF16OUT)
                ((unsigned short*)Cout)[(size_t)row * N + col] = f2bf(acc[t][i]);
            else
                ((float*)Cout)[(size_t)row * N + col] = acc[t][i];
        }
    }
}

// ---------------------------------------------------------------------------
// RoPE in-place on bf16 Q [S,32,128] and K [S,8,128]
// out[i]    = x[i]*cos - x[i+64]*sin ;  out[i+64] = x[i+64]*cos + x[i]*sin
// ---------------------------------------------------------------------------
__global__ void rope_bf16(unsigned short* __restrict__ Q,
                          unsigned short* __restrict__ K, int S)
{
    int idx = blockIdx.x * blockDim.x + threadIdx.x;
    int totq = S * NUM_HEADS * 64;
    int tot  = totq + S * NUM_KV * 64;
    if (idx >= tot) return;

    unsigned short* p;
    int s, head, i, nheads;
    if (idx < totq) {
        p = Q; nheads = NUM_HEADS;
        i = idx & 63; int t = idx >> 6;
        head = t & (NUM_HEADS - 1); s = t >> 5;
    } else {
        p = K; nheads = NUM_KV;
        int t = idx - totq;
        i = t & 63; t >>= 6;
        head = t & (NUM_KV - 1); s = t >> 3;
    }
    float ang = (float)s * __powf(10000.0f, -(float)(2 * i) * (1.0f / 128.0f));
    float c, sn;
    __sincosf(ang, &sn, &c);
    size_t off = ((size_t)s * nheads + head) * HEAD_DIM;
    float x1 = bf2f(p[off + i]);
    float x2 = bf2f(p[off + 64 + i]);
    p[off + i]      = f2bf(x1 * c - x2 * sn);
    p[off + 64 + i] = f2bf(x2 * c + x1 * sn);
}

// ---------------------------------------------------------------------------
// Flash attention (non-causal, GQA group=4).
// Block = (head, 64 query rows), 128 threads = 4 waves, wave = 16 q rows.
// Key tiles of 32; QK^T and PV both via WMMA bf16.
// Q and K tiles staged by the Tensor Data Mover when available.
// ---------------------------------------------------------------------------
__global__ __launch_bounds__(128) void attn_fwd(
    const unsigned short* __restrict__ Qb,   // [S, 4096]
    const unsigned short* __restrict__ Kb,   // [S, 1024]
    const unsigned short* __restrict__ Vb,   // [S, 1024]
    unsigned short* __restrict__ Ob,         // [S, 4096]
    int S)
{
    __shared__ __align__(16) unsigned short q_lds[64 * HEAD_DIM];   // [qrow][d]
    __shared__ __align__(16) unsigned short k_lds[32 * HEAD_DIM];   // [key][d]
    __shared__ __align__(16) unsigned short vt_lds[HEAD_DIM * 32];  // [d][key]
    __shared__ __align__(16) unsigned short p_lds[4][16 * 32];      // per-wave P

    const int h    = blockIdx.x;
    const int qt   = blockIdx.y;
    const int kvh  = h >> 2;                  // 32 heads / 8 kv heads
    const int tid  = threadIdx.x;
    const int wave = tid >> 5;
    const int lane = tid & 31;
    const int hi   = lane >> 4;
    const int l16  = lane & 15;
    const float scale = 0.08838834764831845f; // 1/sqrt(128)

    // --- stage Q tile once: 64x128 bf16 ---
#ifdef HAVE_TDM
    if (wave == 0) {
        tdm_load_2d_bf16((unsigned)(uintptr_t)q_lds,
                         Qb + (size_t)(qt * 64) * HSZ + h * HEAD_DIM,
                         HEAD_DIM, 64, HSZ);
        wait_tensorcnt0();
    }
#else
    {
        int r = tid >> 1, c = (tid & 1) * 64;
        const unsigned* qp = (const unsigned*)(Qb + (size_t)(qt * 64 + r) * HSZ + h * HEAD_DIM + c);
        unsigned* dp = (unsigned*)(q_lds + r * HEAD_DIM + c);
#pragma unroll
        for (int j = 0; j < 32; ++j) dp[j] = qp[j];
    }
#endif

    v8f   o[8] = {};
    float mrow[8], lrow[8];
#pragma unroll
    for (int i = 0; i < 8; ++i) { mrow[i] = -1.0e30f; lrow[i] = 0.0f; }

    const int nkt = S / 32;
    for (int kt = 0; kt < nkt; ++kt) {
        __syncthreads();   // previous tile fully consumed before overwrite
        // --- stage K tile 32x128 row-major ---
#ifdef HAVE_TDM
        if (wave == 0)
            tdm_load_2d_bf16((unsigned)(uintptr_t)k_lds,
                             Kb + (size_t)(kt * 32) * KVSZ + kvh * HEAD_DIM,
                             HEAD_DIM, 32, KVSZ);
#else
        {
            int r = tid >> 2, c = (tid & 3) * 32;
            const unsigned* kp = (const unsigned*)(Kb + (size_t)(kt * 32 + r) * KVSZ + kvh * HEAD_DIM + c);
            unsigned* dp = (unsigned*)(k_lds + r * HEAD_DIM + c);
#pragma unroll
            for (int j = 0; j < 16; ++j) dp[j] = kp[j];
        }
#endif
        // --- stage V tile transposed: vt[d][key] (TDM cannot transpose) ---
        {
            int r = tid >> 2, c = (tid & 3) * 32;
            const unsigned* vp = (const unsigned*)(Vb + (size_t)(kt * 32 + r) * KVSZ + kvh * HEAD_DIM + c);
#pragma unroll
            for (int j = 0; j < 16; ++j) {
                unsigned d = vp[j];
                int dd = c + 2 * j;
                vt_lds[dd * 32 + r]       = (unsigned short)(d & 0xFFFFu);
                vt_lds[(dd + 1) * 32 + r] = (unsigned short)(d >> 16);
            }
        }
#ifdef HAVE_TDM
        if (wave == 0) wait_tensorcnt0();
#endif
        __syncthreads();

        // --- scores: S[16 x 32] = Q[16x128] . K^T, 4 k-steps x 2 key tiles ---
        v8f s0 = {}, s1 = {};
#pragma unroll
        for (int dc = 0; dc < 4; ++dc) {
            Frag aq;
#pragma unroll
            for (int j = 0; j < 8; ++j)
                aq.u[j] = *(const unsigned*)(q_lds + (wave * 16 + l16) * HEAD_DIM + dc * 32 + kkA(j, hi));
            Frag bk0, bk1;
#pragma unroll
            for (int j = 0; j < 8; ++j) {
                bk0.u[j] = *(const unsigned*)(k_lds + l16 * HEAD_DIM        + dc * 32 + kkB(j, hi));
                bk1.u[j] = *(const unsigned*)(k_lds + (16 + l16) * HEAD_DIM + dc * 32 + kkB(j, hi));
            }
            s0 = __builtin_amdgcn_wmma_f32_16x16x32_bf16(false, aq.v, false, bk0.v, (short)0, s0, false, false);
            s1 = __builtin_amdgcn_wmma_f32_16x16x32_bf16(false, aq.v, false, bk1.v, (short)0, s1, false, false);
        }

        // --- online softmax (rows live in 16-lane halves) ---
        float corr[8];
#pragma unroll
        for (int i = 0; i < 8; ++i) {
            float a = s0[i] * scale, b = s1[i] * scale;
            float x = fmaxf(a, b);
#pragma unroll
            for (int off = 8; off >= 1; off >>= 1)
                x = fmaxf(x, __shfl_xor(x, off, 16));
            float mn = fmaxf(mrow[i], x);
            corr[i]  = __expf(mrow[i] - mn);
            float p0 = __expf(a - mn);
            float p1 = __expf(b - mn);
            float rs = p0 + p1;
#pragma unroll
            for (int off = 8; off >= 1; off >>= 1)
                rs += __shfl_xor(rs, off, 16);
            lrow[i] = lrow[i] * corr[i] + rs;
            mrow[i] = mn;
            int r = i + hi * 8;
            p_lds[wave][r * 32 + l16]      = f2bf(p0);
            p_lds[wave][r * 32 + 16 + l16] = f2bf(p1);
        }

        // --- rescale O, then O += P @ V ---
#pragma unroll
        for (int t = 0; t < 8; ++t)
#pragma unroll
            for (int i = 0; i < 8; ++i)
                o[t][i] *= corr[i];

        Frag ap;
#pragma unroll
        for (int j = 0; j < 8; ++j)
            ap.u[j] = *(const unsigned*)(&p_lds[wave][l16 * 32 + kkA(j, hi)]);

#pragma unroll
        for (int t = 0; t < 8; ++t) {
            Frag bv;
#pragma unroll
            for (int j = 0; j < 8; ++j)
                bv.u[j] = *(const unsigned*)(vt_lds + (t * 16 + l16) * 32 + kkB(j, hi));
            o[t] = __builtin_amdgcn_wmma_f32_16x16x32_bf16(false, ap.v, false, bv.v, (short)0, o[t], false, false);
        }
    }

    // --- normalize and store bf16 [s, h*128 + d] ---
#pragma unroll
    for (int t = 0; t < 8; ++t)
#pragma unroll
        for (int i = 0; i < 8; ++i) {
            int row = qt * 64 + wave * 16 + i + hi * 8;
            int col = h * HEAD_DIM + t * 16 + l16;
            Ob[(size_t)row * HSZ + col] = f2bf(o[t][i] / lrow[i]);
        }
}

// ---------------------------------------------------------------------------
// Host-side launch
// ---------------------------------------------------------------------------
extern "C" void kernel_launch(void* const* d_in, const int* in_sizes, int n_in,
                              void* d_out, int out_size, void* d_ws, size_t ws_size,
                              hipStream_t stream) {
    const float* hs = (const float*)d_in[0];
    const float* Wq = (const float*)d_in[1];
    const float* Wk = (const float*)d_in[2];
    const float* Wv = (const float*)d_in[3];
    const float* Wo = (const float*)d_in[4];

    const int S = in_sizes[0] / HSZ;   // 2048

    // workspace carve-up (bf16 buffers)
    char* ws = (char*)d_ws;
    size_t off = 0;
    auto carve = [&](size_t bytes) -> char* {
        char* p = ws + off;
        off = (off + bytes + 255) & ~(size_t)255;
        return p;
    };
    unsigned short* xb  = (unsigned short*)carve((size_t)S * HSZ * 2);
    unsigned short* wqb = (unsigned short*)carve((size_t)HSZ * HSZ * 2);
    unsigned short* wkb = (unsigned short*)carve((size_t)HSZ * KVSZ * 2);
    unsigned short* wvb = (unsigned short*)carve((size_t)HSZ * KVSZ * 2);
    unsigned short* wob = (unsigned short*)carve((size_t)HSZ * HSZ * 2);
    unsigned short* qb  = (unsigned short*)carve((size_t)S * HSZ * 2);
    unsigned short* kb  = (unsigned short*)carve((size_t)S * KVSZ * 2);
    unsigned short* vb  = (unsigned short*)carve((size_t)S * KVSZ * 2);
    unsigned short* ab  = (unsigned short*)carve((size_t)S * HSZ * 2);
    (void)ws_size; (void)n_in; (void)out_size;

    auto cvt = [&](const float* s, unsigned short* d, int n) {
        cvt_f32_bf16<<<(n + 255) / 256, 256, 0, stream>>>(s, d, n);
    };
    cvt(hs, xb, S * HSZ);
    cvt(Wq, wqb, HSZ * HSZ);
    cvt(Wk, wkb, HSZ * KVSZ);
    cvt(Wv, wvb, HSZ * KVSZ);
    cvt(Wo, wob, HSZ * HSZ);

    // Q/K/V projections (bf16 out)
    gemm_bf16<true><<<dim3(HSZ / 128, S / 64), 256, 0, stream>>>(xb, wqb, qb, S, HSZ, HSZ);
    gemm_bf16<true><<<dim3(KVSZ / 128, S / 64), 256, 0, stream>>>(xb, wkb, kb, S, KVSZ, HSZ);
    gemm_bf16<true><<<dim3(KVSZ / 128, S / 64), 256, 0, stream>>>(xb, wvb, vb, S, KVSZ, HSZ);

    // RoPE on Q and K
    {
        int tot = S * NUM_HEADS * 64 + S * NUM_KV * 64;
        rope_bf16<<<(tot + 255) / 256, 256, 0, stream>>>(qb, kb, S);
    }

    // attention
    attn_fwd<<<dim3(NUM_HEADS, S / 64), 128, 0, stream>>>(qb, kb, vb, ab, S);

    // output projection (fp32 out)
    gemm_bf16<false><<<dim3(HSZ / 128, S / 64), 256, 0, stream>>>(ab, wob, (float*)d_out, S, HSZ, HSZ);
}
